// tokenWindowAttention_25340307046829
// MI455X (gfx1250) — matrix operand order repaired and checked by reading, not compiled, not run on hardware
//
#include <hip/hip_runtime.h>

// ---------------------------------------------------------------------------
// Fused windowed attention for MI455X (gfx1250), wave32 + v_wmma_f32_16x16x32_bf16
// One workgroup (8 wave32) per batch window; all intermediates live in LDS
// (258.6 KB of the WGP's 320 KB).
// ---------------------------------------------------------------------------

#define QN    53          // query tokens (4 prompts + 49 window)
#define KVN   49          // key/value tokens
#define NP    4           // num prompts
#define DIMX  192
#define HEADS 6
#define HD    32
#define WIN   7

#define XROWS 68          // x staged rows: 64 kv-aligned + NP shift headroom
#define LDX   200         // bf16 ld for x / q / o buffers (400B rows, 16B aligned)
#define LDW   200         // bf16 ld for transposed weight staging
#define LDK   40          // bf16 ld for k[h][row][d]
#define LDV   72          // bf16 ld for vT[h][d][row]
#define LDA   68          // f32 ld for attention scores (272B rows, bank-skewed)

#define SCALE_F 0.17677669529663687f   // 32^-0.5

// ---- static LDS layout (bytes) --------------------------------------------
#define OFF_X 0
#define SZ_X  (XROWS * LDX * 2)                 // 27200   x (reused as o)
#define OFF_W (OFF_X + SZ_X)                    // 27200
#define SZ_W  (2 * DIMX * LDW * 2)              // 153600  Wt / attnF(104448) / projWt
#define OFF_Q (OFF_W + SZ_W)                    // 180800
#define SZ_Q  (64 * LDX * 2)                    // 25600   q bf16
#define OFF_K (OFF_Q + SZ_Q)                    // 206400
#define SZ_K  (HEADS * 64 * LDK * 2)            // 30720   k bf16
#define OFF_V (OFF_K + SZ_K)                    // 237120
#define SZ_V  (HEADS * HD * LDV * 2)            // 27648   vT bf16
#define SMEM_BYTES (OFF_V + SZ_V)               // 264768
static_assert(SMEM_BYTES <= 320 * 1024, "LDS over WGP budget");

typedef __attribute__((ext_vector_type(16))) __bf16 bf16x16;
typedef __attribute__((ext_vector_type(8)))  __bf16 bf16x8;
typedef __attribute__((ext_vector_type(2)))  __bf16 bf16x2;
typedef __attribute__((ext_vector_type(8)))  float  f32x8;

// --- fp32 -> bf16 via native fptrunc: gfx1250 has HW bf16 converts ---------
__device__ __forceinline__ unsigned pkbf(float a, float b) {
  union { bf16x2 h; unsigned u; } x;
  x.h[0] = (__bf16)a;
  x.h[1] = (__bf16)b;
  return x.u;
}

__device__ __forceinline__ unsigned short f2bf(float f) {
  union { __bf16 h; unsigned short u; } x;
  x.h = (__bf16)f;
  return x.u;
}

__device__ __forceinline__ bf16x16 cat8(bf16x8 lo, bf16x8 hi) {
  bf16x16 r;
#pragma unroll
  for (int i = 0; i < 8; ++i) { r[i] = lo[i]; r[8 + i] = hi[i]; }
  return r;
}

// A fragment (16x32 bf16): lane l16 = row, chunks [kk+half*8, +8) and +16.
__device__ __forceinline__ bf16x16 frag_a_bf(const unsigned short* base, int ld,
                                             int row0, int kk, int l16, int half) {
  const unsigned short* p = base + (row0 + l16) * ld + kk + half * 8;
  return cat8(*(const bf16x8*)p, *(const bf16x8*)(p + 16));
}

// B fragment (32x16 bf16): lane l16 = column n, chunks [kk+half*8, +8) and +16.
__device__ __forceinline__ bf16x16 frag_b_bf(const unsigned short* base, int ld,
                                             int n0, int kk, int l16, int half) {
  const unsigned short* p = base + (n0 + l16) * ld + kk + half * 8;
  return cat8(*(const bf16x8*)p, *(const bf16x8*)(p + 16));
}

// A fragment sourced from f32 LDS (softmax probs), converted to bf16.
__device__ __forceinline__ bf16x16 frag_a_f32(const float* base, int ld,
                                              int row0, int kk, int l16, int half) {
  const float* p = base + (row0 + l16) * ld + kk + half * 8;
  bf16x16 r;
#pragma unroll
  for (int i = 0; i < 8; ++i) {
    r[i]     = (__bf16)p[i];
    r[i + 8] = (__bf16)p[i + 16];
  }
  return r;
}

__device__ __forceinline__ f32x8 wmma_bf(bf16x16 a, bf16x16 b, f32x8 c) {
  return __builtin_amdgcn_wmma_f32_16x16x32_bf16(false, a, false, b,
                                                 (short)0, c, false, false);
}

__global__ __launch_bounds__(256, 1)
void win_attn_wmma_kernel(const float* __restrict__ x,
                          const float* __restrict__ mask,
                          const float* __restrict__ q_w,
                          const float* __restrict__ q_b,
                          const float* __restrict__ kv_w,
                          const float* __restrict__ kv_b,
                          const float* __restrict__ proj_w,
                          const float* __restrict__ proj_b,
                          const float* __restrict__ bias_table,
                          float* __restrict__ out) {
  __shared__ __align__(16) unsigned char smem[SMEM_BYTES];
  unsigned short* xb    = (unsigned short*)(smem + OFF_X);   // [XROWS][LDX]
  unsigned short* wt    = (unsigned short*)(smem + OFF_W);   // weight staging
  float*          attnF = (float*)(smem + OFF_W);            // [H][64][LDA]
  unsigned short* qb    = (unsigned short*)(smem + OFF_Q);   // [64][LDX]
  unsigned short* kb    = (unsigned short*)(smem + OFF_K);   // [H][64][LDK]
  unsigned short* vT    = (unsigned short*)(smem + OFF_V);   // [H][HD][LDV]
  unsigned short* ob    = xb;                                // o reuses x

  const int b    = blockIdx.x;
  const int tid  = threadIdx.x;
  const int lane = tid & 31;
  const int wave = tid >> 5;
  const int l16  = lane & 15;
  const int half = lane >> 4;

  // Warm weights into L2 (global_prefetch_b8), shared by all 4096 blocks.
  for (int i = tid; i < 4 * DIMX; i += 256) {
    int m = i / DIMX, r = i % DIMX;
    const float* p = (m == 0) ? (q_w + r * DIMX)
                   : (m == 1) ? (proj_w + r * DIMX)
                   : (kv_w + (r + (m - 2) * 96) * 2 * DIMX);
    __builtin_prefetch(p, 0, 1);
  }

  // -------- Phase 0: x[b] -> LDS bf16, zero-pad rows QN..XROWS-1 -----------
  {
    const float* xs = x + (size_t)b * QN * DIMX;
    for (int i = tid; i < XROWS * DIMX; i += 256) {
      int r = i / DIMX, c = i % DIMX;
      float v = (r < QN) ? xs[r * DIMX + c] : 0.0f;
      xb[r * LDX + c] = f2bf(v);
    }
  }
  __syncthreads();

  // -------- Phase 1a: stage q_w^T ------------------------------------------
  for (int i = tid; i < DIMX * DIMX; i += 256) {
    int k = i / DIMX, n = i % DIMX;
    wt[n * LDW + k] = f2bf(q_w[i]);
  }
  __syncthreads();

  // -------- GEMM1: q = SCALE * (x @ q_w + q_b) -> qb bf16 ------------------
  for (int t = wave; t < 4 * 12; t += 8) {
    int tm = t / 12, tn = t % 12;
    f32x8 acc = {0, 0, 0, 0, 0, 0, 0, 0};
#pragma unroll
    for (int kk = 0; kk < DIMX; kk += 32) {
      bf16x16 af = frag_a_bf(xb, LDX, tm * 16, kk, l16, half);
      bf16x16 bm = frag_b_bf(wt, LDW, tn * 16, kk, l16, half);
      acc = wmma_bf(af, bm, acc);
    }
    int n = tn * 16 + l16;
    float bias = q_b[n];
#pragma unroll
    for (int r = 0; r < 8; r += 2) {
      unsigned pk = pkbf((acc[r] + bias) * SCALE_F, (acc[r + 1] + bias) * SCALE_F);
      int row = tm * 16 + half * 8 + r;
      qb[row * LDX + n]       = (unsigned short)pk;
      qb[(row + 1) * LDX + n] = (unsigned short)(pk >> 16);
    }
  }
  __syncthreads();

  // -------- Phase 1b: stage kv_w^T (384 x 192) -----------------------------
  for (int i = tid; i < DIMX * 2 * DIMX; i += 256) {
    int k = i / (2 * DIMX), n = i % (2 * DIMX);
    wt[n * LDW + k] = f2bf(kv_w[i]);
  }
  __syncthreads();

  // -------- GEMM2: kv = x[NP:] @ kv_w + kv_b -> kb / vT --------------------
  for (int t = wave; t < 4 * 24; t += 8) {
    int tm = t / 24, tn = t % 24;
    f32x8 acc = {0, 0, 0, 0, 0, 0, 0, 0};
#pragma unroll
    for (int kk = 0; kk < DIMX; kk += 32) {
      bf16x16 af = frag_a_bf(xb + NP * LDX, LDX, tm * 16, kk, l16, half);
      bf16x16 bm = frag_b_bf(wt, LDW, tn * 16, kk, l16, half);
      acc = wmma_bf(af, bm, acc);
    }
    int n = tn * 16 + l16;          // 0..383 ; tn<12 -> k, tn>=12 -> v (uniform)
    float bias = kv_b[n];
    if (tn < 12) {
      int h = n / HD, d = n % HD;
      unsigned short* kp = kb + h * 64 * LDK + d;
#pragma unroll
      for (int r = 0; r < 8; r += 2) {
        unsigned pk = pkbf(acc[r] + bias, acc[r + 1] + bias);
        int row = tm * 16 + half * 8 + r;
        kp[row * LDK]       = (unsigned short)pk;
        kp[(row + 1) * LDK] = (unsigned short)(pk >> 16);
      }
    } else {
      int h = (n - DIMX) / HD, d = (n - DIMX) % HD;
      unsigned short* vp = vT + h * HD * LDV + d * LDV;
#pragma unroll
      for (int r = 0; r < 8; r += 2) {
        unsigned pk = pkbf(acc[r] + bias, acc[r + 1] + bias);
        int row = tm * 16 + half * 8 + r;
        vp[row]     = (unsigned short)pk;
        vp[row + 1] = (unsigned short)(pk >> 16);
      }
    }
  }
  __syncthreads();

  // -------- GEMM3: scores = q @ k^T -> attnF f32 (per head, K=32) ----------
  for (int t = wave; t < HEADS * 16; t += 8) {
    int h = t / 16, tt = t % 16, tm = tt / 4, tn = tt % 4;
    f32x8 acc = {0, 0, 0, 0, 0, 0, 0, 0};
    bf16x16 af = frag_a_bf(qb, LDX, tm * 16, h * HD, l16, half);
    bf16x16 bm = frag_b_bf(kb + h * 64 * LDK, LDK, tn * 16, 0, l16, half);
    acc = wmma_bf(af, bm, acc);
    float* ap = attnF + h * 64 * LDA;
#pragma unroll
    for (int r = 0; r < 8; ++r)
      ap[(tm * 16 + half * 8 + r) * LDA + tn * 16 + l16] = acc[r];
  }
  __syncthreads();

  // -------- Softmax + relative-position bias + window mask -----------------
  {
    const float* mrow = mask + (size_t)(b & 63) * KVN * KVN;
    for (int rid = tid; rid < HEADS * QN; rid += 256) {
      int h = rid / QN, qi = rid % QN;
      float* ap = attnF + h * 64 * LDA + qi * LDA;
      int i = qi - NP;
      float mx = -1e30f;
      for (int j = 0; j < KVN; ++j) {
        float s = ap[j];
        if (i >= 0) {
          int ri = i / WIN, ci = i % WIN, rj = j / WIN, cj = j % WIN;
          int idx = (ri - rj + WIN - 1) * (2 * WIN - 1) + (ci - cj + WIN - 1);
          s += bias_table[idx * HEADS + h];
          s += mrow[i * KVN + j];
        }
        ap[j] = s;
        mx = fmaxf(mx, s);
      }
      float sum = 0.0f;
      for (int j = 0; j < KVN; ++j) {
        float e = __expf(ap[j] - mx);
        ap[j] = e;
        sum += e;
      }
      float inv = 1.0f / sum;
      for (int j = 0; j < KVN; ++j) ap[j] *= inv;
      for (int j = KVN; j < 64; ++j) ap[j] = 0.0f;   // padded keys -> 0 prob
    }
  }
  __syncthreads();

  // -------- GEMM4: o = P @ v -> ob bf16 (per head, K=64) -------------------
  for (int t = wave; t < HEADS * 8; t += 8) {
    int h = t / 8, tt = t % 8, tm = tt / 2, tn = tt % 2;
    f32x8 acc = {0, 0, 0, 0, 0, 0, 0, 0};
#pragma unroll
    for (int kk = 0; kk < 64; kk += 32) {
      bf16x16 af = frag_a_f32(attnF + h * 64 * LDA, LDA, tm * 16, kk, l16, half);
      bf16x16 bm = frag_b_bf(vT + h * HD * LDV, LDV, tn * 16, kk, l16, half);
      acc = wmma_bf(af, bm, acc);
    }
    int d = tn * 16 + l16;
    unsigned short* op = ob + h * HD + d;
#pragma unroll
    for (int r = 0; r < 8; r += 2) {
      unsigned pk = pkbf(acc[r], acc[r + 1]);
      int row = tm * 16 + half * 8 + r;
      op[row * LDX]       = (unsigned short)pk;
      op[(row + 1) * LDX] = (unsigned short)(pk >> 16);
    }
  }
  __syncthreads();

  // -------- Phase 5: stage proj_w^T, GEMM5, store --------------------------
  for (int i = tid; i < DIMX * DIMX; i += 256) {
    int k = i / DIMX, n = i % DIMX;
    wt[n * LDW + k] = f2bf(proj_w[i]);
  }
  __syncthreads();

  float* outp = out + (size_t)b * QN * DIMX;
  for (int t = wave; t < 4 * 12; t += 8) {
    int tm = t / 12, tn = t % 12;
    f32x8 acc = {0, 0, 0, 0, 0, 0, 0, 0};
#pragma unroll
    for (int kk = 0; kk < DIMX; kk += 32) {
      bf16x16 af = frag_a_bf(ob, LDX, tm * 16, kk, l16, half);
      bf16x16 bm = frag_b_bf(wt, LDW, tn * 16, kk, l16, half);
      acc = wmma_bf(af, bm, acc);
    }
    int n = tn * 16 + l16;
    float bias = proj_b[n];
#pragma unroll
    for (int r = 0; r < 8; ++r) {
      int row = tm * 16 + half * 8 + r;
      if (row < QN) outp[row * DIMX + n] = acc[r] + bias;
    }
  }
}

extern "C" void kernel_launch(void* const* d_in, const int* in_sizes, int n_in,
                              void* d_out, int out_size, void* d_ws, size_t ws_size,
                              hipStream_t stream) {
  const float* x          = (const float*)d_in[0];
  const float* mask       = (const float*)d_in[1];
  const float* q_w        = (const float*)d_in[2];
  const float* q_b        = (const float*)d_in[3];
  const float* kv_w       = (const float*)d_in[4];
  const float* kv_b       = (const float*)d_in[5];
  const float* proj_w     = (const float*)d_in[6];
  const float* proj_b     = (const float*)d_in[7];
  const float* bias_table = (const float*)d_in[8];
  float* outp             = (float*)d_out;

  const int B = in_sizes[0] / (QN * DIMX);   // 4096
  hipLaunchKernelGGL(win_attn_wmma_kernel, dim3(B), dim3(256), 0, stream,
                     x, mask, q_w, q_b, kv_w, kv_b, proj_w, proj_b, bias_table,
                     outp);
}
